// BlockFourierCirculant_82849919140117
// MI455X (gfx1250) — compile-verified, matrix operand order checked by reading, not run on hardware
//
#include <hip/hip_runtime.h>

// ---------------------------------------------------------------------------
// BlockFourierCirculant on MI455X (gfx1250).
// y = x @ B with B the dense block-circulant matrix equivalent to
// rfft -> per-bin 16x16 complex mix -> irfft on 256-wide blocks.
//
// Split-bf16 (hi+lo) GEMM on v_wmma_f32_16x16x32_bf16, with BOTH operands
// prepacked in global memory in the exact per-lane WMMA fragment layout:
//   tile(16 rows x 32 K) -> [hi(32 lanes x 16 bf16)][lo(32 lanes x 16 bf16)]
//   = 2048 bytes per tile pair, lane-contiguous 32B chunks.
// GEMM staging is then a byte-exact linear copy -> global_load_async_to_lds
// (ASYNCcnt) double-buffered; fragments load as ds_load_b128 pairs.
//
// Workspace layout (needs ~192.25 MB):
//   [0, 256KB)                  float  c[16][16][256]      irfft(W) columns
//   [256KB, +64MB)              packed B  (256 nt x 128 kt x 2KB)
//   [256KB+64MB, +128MB)        packed Xh/Xl (512 mt x 128 kt x 2KB)
// ---------------------------------------------------------------------------

typedef __attribute__((ext_vector_type(16))) __bf16 v16bf;
typedef __attribute__((ext_vector_type(8)))  float  v8f;

#define NFEAT  4096
#define MBATCH 8192
#define KHALF  129

#define C_OFS   0
#define BP_OFS  262144                      // 256KB
#define XP_OFS  (262144 + 67108864)         // 256KB + 64MB

static __device__ __forceinline__ unsigned short f2bf(float f) {
  unsigned u = __float_as_uint(f);
  unsigned r = 0x7FFFu + ((u >> 16) & 1u);   // round to nearest even
  return (unsigned short)((u + r) >> 16);
}
static __device__ __forceinline__ float bf2f(unsigned short h) {
  return __uint_as_float(((unsigned)h) << 16);
}
// ISA 16-bit A-operand K mapping: element e, lane-half h -> K offset in tile
static __device__ __forceinline__ int koff(int e, int h) {
  return e + 8 * h + ((e >= 8) ? 8 : 0);
}

// ---- Kernel 1: c[o][i][r] = irfft_256(Wf[o,i])[r] --------------------------
__global__ void build_c_kernel(const float* __restrict__ Wr,
                               const float* __restrict__ Wi,
                               float* __restrict__ c) {
  int idx = blockIdx.x * blockDim.x + threadIdx.x;     // 0..65535
  int r = idx & 255;
  int oi = idx >> 8;
  const float* wr = Wr + (size_t)oi * KHALF;
  const float* wi = Wi + (size_t)oi * KHALF;
  float acc = wr[0];
  acc += (r & 1) ? -wr[128] : wr[128];
  const float w0 = 6.28318530717958647692f / 256.0f;
  #pragma unroll 4
  for (int k = 1; k < 128; ++k) {
    int ph = (k * r) & 255;                            // exact phase mod 256
    float s, cs;
    __sincosf(w0 * (float)ph, &s, &cs);
    acc += 2.0f * (wr[k] * cs - wi[k] * s);
  }
  c[idx] = acc * (1.0f / 256.0f);
}

// ---- Kernel 2: packed split-bf16 circulant matrix B ------------------------
// One thread per (nt, kt, lane): emits 16 hi + 16 lo bf16 (32B + 32B).
__global__ void build_Bp_kernel(const float* __restrict__ c,
                                unsigned short* __restrict__ Bp) {
  int idx = blockIdx.x * blockDim.x + threadIdx.x;     // 0..1048575
  int lane = idx & 31;
  int rest = idx >> 5;
  int kt = rest & 127;
  int nt = rest >> 7;                                  // 0..255
  int r = lane & 15, h = lane >> 4;
  int n = nt * 16 + r;
  int o = n >> 8, nn = n & 255;

  union { unsigned short s[16]; uint4 q[2]; } hi, lo;
  #pragma unroll
  for (int e = 0; e < 16; ++e) {
    int k = kt * 32 + koff(e, h);
    int i = k >> 8, t = k & 255;
    float v = c[((o * 16 + i) << 8) + ((nn - t) & 255)];
    unsigned short hb = f2bf(v);
    hi.s[e] = hb;
    lo.s[e] = f2bf(v - bf2f(hb));
  }
  // layout: ((nt*128 + kt) tile pair) * 1024 ushorts; hi then lo, lane*16 each
  unsigned short* d = Bp + (size_t)(nt * 128 + kt) * 1024 + lane * 16;
  *(uint4*)(d)        = hi.q[0];
  *(uint4*)(d + 8)    = hi.q[1];
  *(uint4*)(d + 512)  = lo.q[0];
  *(uint4*)(d + 520)  = lo.q[1];
}

// ---- Kernel 3: packed split-bf16 activations -------------------------------
__global__ void xsplit_kernel(const float* __restrict__ X,
                              unsigned short* __restrict__ Xp) {
  int idx = blockIdx.x * blockDim.x + threadIdx.x;     // 0..2097151
  int lane = idx & 31;
  int rest = idx >> 5;
  int kt = rest & 127;
  int mt = rest >> 7;                                  // 0..511
  int r = lane & 15, h = lane >> 4;
  const float* src = X + (size_t)(mt * 16 + r) * NFEAT + kt * 32;

  union { unsigned short s[16]; uint4 q[2]; } hi, lo;
  #pragma unroll
  for (int e = 0; e < 16; ++e) {
    float v = src[koff(e, h)];
    unsigned short hb = f2bf(v);
    hi.s[e] = hb;
    lo.s[e] = f2bf(v - bf2f(hb));
  }
  unsigned short* d = Xp + (size_t)(mt * 128 + kt) * 1024 + lane * 16;
  *(uint4*)(d)        = hi.q[0];
  *(uint4*)(d + 8)    = hi.q[1];
  *(uint4*)(d + 512)  = lo.q[0];
  *(uint4*)(d + 520)  = lo.q[1];
}

// ---- Kernel 4: GEMM, async-staged, double-buffered, WMMA bf16 split --------
__global__ __launch_bounds__(256) void gemm_kernel(
    const char* __restrict__ Xp,   // packed activations (byte base)
    const char* __restrict__ Bp,   // packed B (byte base)
    float* __restrict__ Y) {
  // two buffers x (A:16KB + B:16KB)
  __shared__ __align__(32) char smem[65536];

  const int tid  = threadIdx.x;
  const int wave = tid >> 5;
  const int lane = tid & 31;
  const int wm = wave & 3;            // wave M offset = wm*32 (2 m-tiles)
  const int wn = wave >> 2;           // wave N offset = wn*64 (4 n-tiles)
  const int mt0 = blockIdx.x * 8;     // 8 m-tiles per block (BM=128)
  const int nt0 = blockIdx.y * 8;     // 8 n-tiles per block (BN=128)
  const int half = lane >> 4;
  const int r    = lane & 15;

  const unsigned lds0 = (unsigned)(uintptr_t)(void*)smem;  // LDS byte base

  // Loop-invariant per-thread copy addresses. Per k-step: A region = 8 tile
  // pairs * 2KB = 16KB, same for B. Each thread copies 4x16B per region.
  unsigned voffA[4], voffB[4], ldsA[4], ldsB[4];
  #pragma unroll
  for (int j = 0; j < 4; ++j) {
    unsigned b = (unsigned)(tid + j * 256) * 16u;     // byte id in region
    unsigned tile = b >> 11;                          // 0..7
    unsigned within = b & 2047u;
    voffA[j] = ((mt0 + tile) << 18) + within;         // + kt*2048 via saddr
    voffB[j] = ((nt0 + tile) << 18) + within;
    ldsA[j] = lds0 + b;
    ldsB[j] = lds0 + 16384u + b;
  }

  v8f acc[2][4];
  #pragma unroll
  for (int i2 = 0; i2 < 2; ++i2)
    #pragma unroll
    for (int j2 = 0; j2 < 4; ++j2)
      acc[i2][j2] = (v8f){0.f, 0.f, 0.f, 0.f, 0.f, 0.f, 0.f, 0.f};

  union Frag { v16bf v; uint4 q[2]; };

  const char* sA = Xp;                // advances 2KB per k-step
  const char* sB = Bp;

  // prologue: stage k-step 0 into buffer 0
  #pragma unroll
  for (int j = 0; j < 4; ++j) {
    asm volatile("global_load_async_to_lds_b128 %0, %1, %2"
                 :: "v"(ldsA[j]), "v"(voffA[j]), "s"(sA) : "memory");
    asm volatile("global_load_async_to_lds_b128 %0, %1, %2"
                 :: "v"(ldsB[j]), "v"(voffB[j]), "s"(sB) : "memory");
  }
  asm volatile("s_wait_asynccnt 0x0" ::: "memory");
  __syncthreads();

  for (int s = 0; s < 128; ++s) {
    const int cur = s & 1;

    // issue async staging for step s+1 into the other buffer (overlaps WMMA)
    if (s < 127) {
      sA += 2048; sB += 2048;
      const unsigned bofs = (unsigned)(1 - cur) << 15;
      #pragma unroll
      for (int j = 0; j < 4; ++j) {
        asm volatile("global_load_async_to_lds_b128 %0, %1, %2"
                     :: "v"(ldsA[j] + bofs), "v"(voffA[j]), "s"(sA) : "memory");
        asm volatile("global_load_async_to_lds_b128 %0, %1, %2"
                     :: "v"(ldsB[j] + bofs), "v"(voffB[j]), "s"(sB) : "memory");
      }
    }

    // fragment loads: lane-contiguous 32B -> 2x ds_load_b128 each
    const char* buf = smem + ((unsigned)cur << 15);
    Frag ah[2], al[2], bh[4], bl[4];
    #pragma unroll
    for (int i2 = 0; i2 < 2; ++i2) {
      const char* p = buf + (wm * 2 + i2) * 2048 + lane * 32;
      ah[i2].q[0] = *(const uint4*)(p);
      ah[i2].q[1] = *(const uint4*)(p + 16);
      al[i2].q[0] = *(const uint4*)(p + 1024);
      al[i2].q[1] = *(const uint4*)(p + 1040);
    }
    #pragma unroll
    for (int j2 = 0; j2 < 4; ++j2) {
      const char* p = buf + 16384 + (wn * 4 + j2) * 2048 + lane * 32;
      bh[j2].q[0] = *(const uint4*)(p);
      bh[j2].q[1] = *(const uint4*)(p + 16);
      bl[j2].q[0] = *(const uint4*)(p + 1024);
      bl[j2].q[1] = *(const uint4*)(p + 1040);
    }

    // split-bf16 accumulation: xh*Bh + xh*Bl + xl*Bh (fp32 accumulate)
    #pragma unroll
    for (int i2 = 0; i2 < 2; ++i2) {
      #pragma unroll
      for (int j2 = 0; j2 < 4; ++j2) {
        acc[i2][j2] = __builtin_amdgcn_wmma_f32_16x16x32_bf16(
            false, ah[i2].v, false, bh[j2].v, (short)0, acc[i2][j2], false, false);
        acc[i2][j2] = __builtin_amdgcn_wmma_f32_16x16x32_bf16(
            false, ah[i2].v, false, bl[j2].v, (short)0, acc[i2][j2], false, false);
        acc[i2][j2] = __builtin_amdgcn_wmma_f32_16x16x32_bf16(
            false, al[i2].v, false, bh[j2].v, (short)0, acc[i2][j2], false, false);
      }
    }

    // next buffer must be fully staged before anyone reads it
    asm volatile("s_wait_asynccnt 0x0" ::: "memory");
    __syncthreads();
  }

  // writeback: C/D layout -> lanes 0-15: M=v (+8 for lanes 16-31), N=lane&15
  #pragma unroll
  for (int i2 = 0; i2 < 2; ++i2) {
    #pragma unroll
    for (int j2 = 0; j2 < 4; ++j2) {
      int gm = blockIdx.x * 128 + wm * 32 + i2 * 16 + 8 * half;
      int gn = blockIdx.y * 128 + wn * 64 + j2 * 16 + r;
      #pragma unroll
      for (int v = 0; v < 8; ++v) {
        Y[(size_t)(gm + v) * NFEAT + gn] = acc[i2][j2][v];
      }
    }
  }
}

extern "C" void kernel_launch(void* const* d_in, const int* in_sizes, int n_in,
                              void* d_out, int out_size, void* d_ws, size_t ws_size,
                              hipStream_t stream) {
  (void)in_sizes; (void)n_in; (void)out_size; (void)ws_size;

  const float* x  = (const float*)d_in[0];
  const float* Wr = (const float*)d_in[1];
  const float* Wi = (const float*)d_in[2];
  float* y = (float*)d_out;

  char* ws = (char*)d_ws;
  float*          c  = (float*)(ws + C_OFS);
  unsigned short* Bp = (unsigned short*)(ws + BP_OFS);
  unsigned short* Xp = (unsigned short*)(ws + XP_OFS);

  // 1) irfft of all 16x16 block spectra -> circulant first columns
  build_c_kernel<<<256, 256, 0, stream>>>(Wr, Wi, c);
  // 2) packed split-bf16 block-circulant B (4096x4096 -> fragment layout)
  build_Bp_kernel<<<4096, 256, 0, stream>>>(c, Bp);
  // 3) packed split-bf16 activations (fragment layout)
  xsplit_kernel<<<8192, 256, 0, stream>>>(x, Xp);
  // 4) y = x * B : async-staged, double-buffered wave32 WMMA GEMM
  dim3 grid(MBATCH / 128, NFEAT / 128);
  gemm_kernel<<<grid, 256, 0, stream>>>((const char*)Bp - BP_OFS + XP_OFS,
                                        (const char*)Bp,
                                        y);
}
